// OneSidedComplexFFT_56049323213334
// MI455X (gfx1250) — matrix operand order repaired and checked by reading, not compiled, not run on hardware
//
#include <hip/hip_runtime.h>

// One-sided 512-pt DFT as a WMMA GEMM:
//   out(64000 x 514) = x(64000 x 512) @ B(512 x 514),  B = interleaved cos/-sin
// Strategy: A register-resident per wave (f32->f16 convert at load, WMMA-A layout),
// B pre-packed into WMMA-B lane layout in d_ws (f16, L2-resident), double-buffered
// in registers so B loads overlap WMMA execution,
// C accumulated in f32 via v_wmma_f32_16x16x32_f16, stored as coalesced {re,im} float2.

typedef __attribute__((ext_vector_type(16))) _Float16 v16h;
typedef __attribute__((ext_vector_type(8)))  float    v8f;

#define FFT_LEN   512
#define NBINS     257            // one-sided bins (K/2+1)
#define NTILE     17             // ceil(257/16) N-tiles of 16 bins (pad to 272)
#define KCHUNKS   16             // 512 / 32 K-steps per WMMA chain
#define M_TILE    128            // rows per workgroup (8 waves x 16 rows)
#define OUT_COLS  514            // 2 * NBINS interleaved re/im

// ---------------------------------------------------------------------------
// B prep: pack real/imag DFT kernels into f16 WMMA-B lane layout.
// Layout (halves): [m(2: re,im)][t(17)][c(16)][lane(32)][h(16)]
//   bin (B column N) = 16*t + (lane & 15)         (0 for padded bins >= 257)
//   n   (B row K)    = 32*c + 8*(lane >> 4) + (h & 7) + 16*(h >> 3)  (B 32x16: lanes 0-15 K=0..7,16..23,
//                                                  lanes 16-31 K=8..15,24..31, 2/VGPR)
// Total = 2*17*16*32*16 = 278528 halves = 557056 bytes of d_ws.
// ---------------------------------------------------------------------------
__global__ __launch_bounds__(256) void dft_prep_b(
    const float* __restrict__ rk,       // (257, 512) cos kernel
    const float* __restrict__ ik,       // (257, 512) -sin kernel
    _Float16* __restrict__ bprep)
{
    const int total = 2 * NTILE * KCHUNKS * 32 * 16;
    int idx = blockIdx.x * blockDim.x + threadIdx.x;
    if (idx >= total) return;

    int h    = idx & 15;
    int rest = idx >> 4;
    int lane = rest & 31;  rest >>= 5;
    int c    = rest & 15;  rest >>= 4;
    int t    = rest % NTILE;
    int m    = rest / NTILE;

    int bin = 16 * t + (lane & 15);
    int n   = 32 * c + 8 * (lane >> 4) + (h & 7) + 16 * (h >> 3);

    float val = 0.0f;
    if (bin < NBINS) {
        const float* __restrict__ src = (m == 0) ? rk : ik;
        val = src[(size_t)bin * FFT_LEN + n];
    }
    bprep[idx] = (_Float16)val;
}

// ---------------------------------------------------------------------------
// Main GEMM kernel. 256 threads = 8 waves, each wave owns 16 rows of the
// 128-row M-tile and keeps its full 16x512 f16 A slab in VGPRs.
// B is double-buffered in registers: loads for K-chunk c+1 are issued before
// the WMMAs consuming chunk c, so the compiler can wait on loadcnt<=2 instead
// of 0 and overlap L0/L2 latency with matrix math.
// ---------------------------------------------------------------------------
__global__ __launch_bounds__(256) void dft_wmma(
    const float*    __restrict__ x,      // (rows, 512) f32
    const _Float16* __restrict__ bprep,  // packed B, layout above
    float*          __restrict__ out)    // (rows, 514) f32 interleaved re/im
{
    const int lane  = threadIdx.x & 31;
    const int wave  = threadIdx.x >> 5;
    const int lmod  = lane & 15;
    const int lhalf = lane >> 4;

    const int rowBase = blockIdx.x * M_TILE + wave * 16;

    // -------- load A tile (16 rows x 512 K) into registers, WMMA-A layout ----
    // Lane L (0-15): row = rowBase+L, holds K {base..base+7, base+16..base+23}
    // Lane L+16   : same row set,    holds K {base+8..base+15, base+24..base+31}
    const int arow = rowBase + lmod;
    const int koff = 8 * lhalf;
    const float* __restrict__ xrow = x + (size_t)arow * FFT_LEN;

    v16h a[KCHUNKS];
#pragma unroll
    for (int c = 0; c < KCHUNKS; ++c) {
        const int base = 32 * c + koff;
        const float4 f0 = *reinterpret_cast<const float4*>(xrow + base);
        const float4 f1 = *reinterpret_cast<const float4*>(xrow + base + 4);
        const float4 f2 = *reinterpret_cast<const float4*>(xrow + base + 16);
        const float4 f3 = *reinterpret_cast<const float4*>(xrow + base + 20);
        v16h av;
        av[0]  = (_Float16)f0.x; av[1]  = (_Float16)f0.y;
        av[2]  = (_Float16)f0.z; av[3]  = (_Float16)f0.w;
        av[4]  = (_Float16)f1.x; av[5]  = (_Float16)f1.y;
        av[6]  = (_Float16)f1.z; av[7]  = (_Float16)f1.w;
        av[8]  = (_Float16)f2.x; av[9]  = (_Float16)f2.y;
        av[10] = (_Float16)f2.z; av[11] = (_Float16)f2.w;
        av[12] = (_Float16)f3.x; av[13] = (_Float16)f3.y;
        av[14] = (_Float16)f3.z; av[15] = (_Float16)f3.w;
        a[c] = av;
    }

    // -------- sweep the 17 N-tiles; real+imag accumulated together ----------
    const v16h* __restrict__ bp = reinterpret_cast<const v16h*>(bprep);
    const int bimOff = NTILE * KCHUNKS * 32;   // v16h units: start of imag half

#pragma unroll 1
    for (int t = 0; t < NTILE; ++t) {
        v8f c_re = {};
        v8f c_im = {};
        const v16h* __restrict__ btile = bp + ((size_t)t * KCHUNKS * 32) + lane;

        // prologue: bring in K-chunk 0
        v16h bre = btile[0];
        v16h bim = btile[bimOff];

#pragma unroll
        for (int c = 0; c < KCHUNKS; ++c) {
            v16h breN, bimN;
            if (c + 1 < KCHUNKS) {           // issue next chunk's loads first
                breN = btile[(c + 1) * 32];
                bimN = btile[(c + 1) * 32 + bimOff];
            }
            c_re = __builtin_amdgcn_wmma_f32_16x16x32_f16(
                       false, a[c], false, bre, (short)0, c_re, false, false);
            c_im = __builtin_amdgcn_wmma_f32_16x16x32_f16(
                       false, a[c], false, bim, (short)0, c_im, false, false);
            if (c + 1 < KCHUNKS) {
                bre = breN;
                bim = bimN;
            }
        }

        // warm the next N-tile's B region while the stores drain
        if (t + 1 < NTILE) {
            __builtin_prefetch(btile + KCHUNKS * 32, 0, 1);
            __builtin_prefetch(btile + KCHUNKS * 32 + bimOff, 0, 1);
        }

        // C layout: VGPR r of lane L holds (M = 8*(L/16)+r, N = L%16).
        // Interleaved {re,im} pair is 8B-contiguous -> coalesced b64 stores.
        const int bin = 16 * t + lmod;
        if (bin < NBINS) {
            float* __restrict__ orow =
                out + (size_t)(rowBase + 8 * lhalf) * OUT_COLS + 2 * bin;
#pragma unroll
            for (int r = 0; r < 8; ++r) {
                float2 v;
                v.x = c_re[r];
                v.y = c_im[r];
                *reinterpret_cast<float2*>(orow + (size_t)r * OUT_COLS) = v;
            }
        }
    }
}

// ---------------------------------------------------------------------------
extern "C" void kernel_launch(void* const* d_in, const int* in_sizes, int n_in,
                              void* d_out, int out_size, void* d_ws, size_t ws_size,
                              hipStream_t stream)
{
    const float* x  = (const float*)d_in[0];   // (16, 4000, 512) f32
    const float* rk = (const float*)d_in[1];   // (257, 512) f32
    const float* ik = (const float*)d_in[2];   // (257, 512) f32
    float* out      = (float*)d_out;           // (16, 4000, 514) f32
    _Float16* bprep = (_Float16*)d_ws;         // needs 557056 bytes

    const int rows = in_sizes[0] / FFT_LEN;    // 64000

    const int prep_total  = 2 * NTILE * KCHUNKS * 32 * 16;   // 278528
    const int prep_blocks = (prep_total + 255) / 256;        // 1088
    dft_prep_b<<<prep_blocks, 256, 0, stream>>>(rk, ik, bprep);

    const int gemm_blocks = rows / M_TILE;                   // 500
    dft_wmma<<<gemm_blocks, 256, 0, stream>>>(x, bprep, out);
}
